// LocalAttention_16956530884876
// MI455X (gfx1250) — matrix operand order repaired and checked by reading, not compiled, not run on hardware
//
#include <hip/hip_runtime.h>
#include <hip/hip_bf16.h>
#include <stdint.h>

// ---------------------------------------------------------------------------
// LocalAttention for MI455X (gfx1250):
//   - all matmuls via v_wmma_f32_16x16x32_bf16 (f32 accumulate)
//   - attention k-tile staged into LDS via the Tensor Data Mover
//     (tensor_load_to_lds + s_wait_tensorcnt), softmax in LDS
// ---------------------------------------------------------------------------

#define BATCH   4
#define NTOK    8192
#define DMODEL  256
#define NHEAD   8
#define INNER   512
#define HD      64
#define WSZ     128
#define JW      256          // (BACK+1)*WS
#define NW      64           // NTOK / WSZ
#define MROWS   (BATCH*NTOK) // 32768

typedef __bf16 bf16_t;
typedef __attribute__((ext_vector_type(16))) __bf16 bf16x16;
typedef __attribute__((ext_vector_type(8)))  float  f32x8;
typedef __attribute__((ext_vector_type(4)))  unsigned int u32x4;
typedef __attribute__((ext_vector_type(8)))  int i32x8;
typedef __attribute__((ext_vector_type(4)))  int i32x4;

__device__ __forceinline__ f32x8 wmma_bf16(bf16x16 a, bf16x16 b, f32x8 c) {
  return __builtin_amdgcn_wmma_f32_16x16x32_bf16(false, a, false, b, (short)0, c,
                                                 false, false);
}

// A-fragment (16x32 bf16) from row-major memory, row stride `ld` elements.
// lane L: row = L%16; elems 0..15 <-> K=kb..kb+15,
// kb = (L>=16)?16:0  -> one contiguous 32B chunk.
__device__ __forceinline__ bf16x16 load_a_frag(const bf16_t* base, int ld) {
  int lane = threadIdx.x & 31;
  int row  = lane & 15;
  int kb   = (lane & 16) ? 16 : 0;
  const bf16_t* p = base + (size_t)row * ld + kb;
  bf16x16 a;
#pragma unroll
  for (int i = 0; i < 16; ++i) { a[i] = p[i]; }
  return a;
}

// B-fragment (32x16 bf16) from memory laid out [n][k] (each output column's
// K values contiguous, stride ldk). lane L: col = L%16, K = (L>=16?16:0)+i.
__device__ __forceinline__ bf16x16 load_b_frag_nk(const bf16_t* base, int ldk) {
  int lane = threadIdx.x & 31;
  int col  = lane & 15;
  int kh   = (lane & 16) ? 16 : 0;
  const bf16_t* p = base + (size_t)col * ldk + kh;
  bf16x16 b;
#pragma unroll
  for (int i = 0; i < 16; ++i) b[i] = p[i];
  return b;
}

// ---------------------------------------------------------------------------
__global__ __launch_bounds__(256) void la_cvt_bf16(const float* __restrict__ s,
                                                   bf16_t* __restrict__ d, int n) {
  int i = blockIdx.x * blockDim.x + threadIdx.x;
  if (i < n) d[i] = (bf16_t)s[i];
}

// ---------------------------------------------------------------------------
// Fused QKV projection: C = X(32768x256) * W^T, W stored (INNER, DMODEL).
// 4 waves / block, each wave -> 16 rows x 64 cols (4 accumulators, 8 K-steps).
// q,k stored (b,h,tok,hd); v stored transposed (b,h,hd,tok).
__global__ __launch_bounds__(128) void la_qkv_gemm(
    const bf16_t* __restrict__ xb,
    const bf16_t* __restrict__ wqb, const bf16_t* __restrict__ wkb,
    const bf16_t* __restrict__ wvb,
    bf16_t* __restrict__ qd, bf16_t* __restrict__ kd, bf16_t* __restrict__ vtd) {
  const int which = blockIdx.z;
  const bf16_t* w = (which == 0) ? wqb : (which == 1) ? wkb : wvb;
  const int wave = threadIdx.x >> 5;
  const int m0 = blockIdx.x * 64 + wave * 16;
  const int e0 = blockIdx.y * 64;

  f32x8 acc[4] = {};
#pragma unroll
  for (int k0 = 0; k0 < DMODEL; k0 += 32) {
    bf16x16 a = load_a_frag(xb + (size_t)m0 * DMODEL + k0, DMODEL);
#pragma unroll
    for (int nt = 0; nt < 4; ++nt) {
      bf16x16 b = load_b_frag_nk(w + (size_t)(e0 + nt * 16) * DMODEL + k0, DMODEL);
      acc[nt] = wmma_bf16(a, b, acc[nt]);
    }
  }

  const int lane = threadIdx.x & 31;
  const int mh = (lane & 16) ? 8 : 0;
  const int nn = lane & 15;
#pragma unroll
  for (int nt = 0; nt < 4; ++nt) {
    int e = e0 + nt * 16 + nn;
    int h = e >> 6, hd = e & 63;
#pragma unroll
    for (int r = 0; r < 8; ++r) {
      int tokg = m0 + mh + r;
      int b = tokg >> 13, tok = tokg & (NTOK - 1);
      bf16_t val = (bf16_t)acc[nt][r];
      size_t bh = (size_t)b * NHEAD + h;
      if (which == 0)      qd[(bh * NTOK + tok) * HD + hd] = val;
      else if (which == 1) kd[(bh * NTOK + tok) * HD + hd] = val;
      else                 vtd[(bh * HD + hd) * NTOK + tok] = val;
    }
  }
}

// ---------------------------------------------------------------------------
// One workgroup (4 waves) per (b,h,window).
// k-tile (256 tokens x 64 hd, bf16, 32KB) is DMA'd into LDS with the TDM.
// Each wave: 16-row q strip, two passes (qh) to cover 128 rows.
// Energy -> LDS (f32) -> masked softmax (also written to d_out attn)
// -> bf16 A-frags -> attn@v WMMA.
__global__ __launch_bounds__(128) void la_attn(
    const bf16_t* __restrict__ qb, const bf16_t* __restrict__ kb,
    const bf16_t* __restrict__ vt, const unsigned char* __restrict__ mask,
    float* __restrict__ attn_out, bf16_t* __restrict__ ob) {
  __shared__ float  sE[4][16 * JW];   // 64 KB softmax strips
  __shared__ bf16_t sK[JW * HD];      // 32 KB k tile (row = window column j)

  const int blk = blockIdx.x;               // b*H*NW + h*NW + w
  const int w = blk & (NW - 1);
  const int h = (blk >> 6) & (NHEAD - 1);
  const int bb = blk >> 9;
  const int wave = threadIdx.x >> 5;
  const int lane = threadIdx.x & 31;
  float* strip = sE[wave];

  const int tw = w * WSZ;
  const int mrow = (bb * NHEAD + h) & (BATCH - 1);   // reference's (b*H+h)%b quirk
  const float scale = 0.0625f;                       // 256^-0.5
  const float NEG = -3.402823466e38f;
  const size_t bh = (size_t)bb * NHEAD + h;
  const int mh = (lane & 16) ? 8 : 0;
  const int kh = (lane & 16) ? 16 : 0;

  // ---- stage k tile into LDS --------------------------------------------
  // window covers global tokens [tw-128, tw+128); tokens < 0 are the
  // reference's -1.0 pad region.
  {
    if (w == 0) {   // pre-fill pad rows (j = 0..127) with -1.0
      for (int i = threadIdx.x; i < WSZ * HD; i += 128)
        sK[i] = (bf16_t)(-1.0f);
    }
    if (threadIdx.x < 32) {  // wave 0 issues the TDM descriptor
      const int tstart = (w == 0) ? 0 : (tw - WSZ);
      const int nrows  = (w == 0) ? WSZ : JW;
      const uint32_t lds_off =
          (uint32_t)(uintptr_t)(&sK[0]) + (uint32_t)((w == 0) ? WSZ * HD * 2 : 0);
      const uint64_t ga =
          (uint64_t)(uintptr_t)(kb + (bh * NTOK + tstart) * HD);
      // D# group 0: count=1 | lds_addr | global_addr(57b) | type=2
      u32x4 g0 = {1u, lds_off, (uint32_t)ga,
                  ((uint32_t)(ga >> 32) & 0x01FFFFFFu) | (2u << 30)};
      // D# group 1: data_size=2B; tensor_dim0=HD, tensor_dim1=nrows;
      //             tile_dim0=HD, tile_dim1=nrows; stride0=HD
      i32x8 g1 = {(int)(1u << 16),                       // data_size=1 (2 bytes)
                  (int)((HD & 0xFFFF) << 16),            // tensor_dim0 lo
                  (int)((nrows & 0xFFFF) << 16),         // dim0 hi | dim1 lo
                  (int)((HD & 0xFFFF) << 16),            // dim1 hi | tile_dim0
                  (int)(nrows & 0xFFFF),                 // tile_dim1 | tile_dim2=0
                  (int)HD,                               // dim0_stride lo
                  0, 0};                                 // stride hi / dim1_stride
      i32x4 gz4 = {0, 0, 0, 0};
      i32x8 gz8 = {0, 0, 0, 0, 0, 0, 0, 0};
      // clang-23 / amdgpu-toolchain 6-arg form (probe-confirmed arity)
      __builtin_amdgcn_tensor_load_to_lds(g0, g1, gz4, gz4, gz8, 0);
      __builtin_amdgcn_s_wait_tensorcnt(0);
    }
    // prefetch v tile (64 rows x 512B, rows strided NTOK*2 bytes) for pass 2
    {
      const int t0c = (w == 0) ? 0 : (tw - WSZ);
      for (int idx = threadIdx.x; idx < 256; idx += 128) {
        int row = idx >> 2, seg = idx & 3;
        __builtin_prefetch(vt + (bh * HD + row) * NTOK + t0c + seg * 64, 0, 0);
      }
    }
    __syncthreads();
  }

  for (int qh = 0; qh < 2; ++qh) {
    const int qi0 = qh * 64 + wave * 16;
    const bf16_t* qbase = qb + (bh * NTOK + tw + qi0) * HD;
    bf16x16 a0 = load_a_frag(qbase + 0, HD);    // d = 0..31
    bf16x16 a1 = load_a_frag(qbase + 32, HD);   // d = 32..63

    unsigned rowm = 0;
#pragma unroll
    for (int r = 0; r < 8; ++r)
      rowm |= (mask[(size_t)mrow * NTOK + tw + qi0 + mh + r] ? 1u : 0u) << r;

    // ---- energy = q k^T, masked, into LDS strip -------------------------
    for (int jt = 0; jt < 16; ++jt) {
      int j = jt * 16 + (lane & 15);     // this lane's column of the D-frag
      int t = tw - WSZ + j;              // global token of k column
      bool pad = (t < 0);
      bool cvalid = !pad && (mask[(size_t)mrow * NTOK + (pad ? 0 : t)] != 0);

      const bf16_t* kp = sK + (size_t)j * HD;   // staged k row (pad rows = -1.0)
      bf16x16 b0, b1;
#pragma unroll
      for (int i = 0; i < 16; ++i) { b0[i] = kp[kh + i]; b1[i] = kp[32 + kh + i]; }

      f32x8 e = {};
      e = wmma_bf16(a0, b0, e);
      e = wmma_bf16(a1, b1, e);

#pragma unroll
      for (int r = 0; r < 8; ++r) {
        int i = qi0 + mh + r;                       // query row in window
        bool qbit = (rowm >> r) & 1;
        bool masked = pad || ((i + WSZ) < j) || !qbit || !cvalid;
        strip[(mh + r) * JW + jt * 16 + (lane & 15)] = masked ? NEG : e[r] * scale;
      }
    }

    // ---- softmax over j (wave-local; 2 lanes per row) -------------------
    {
      int row = lane & 15;
      int ch = (lane & 16) ? 128 : 0;
      float mx = NEG;
      for (int c = 0; c < 128; ++c) mx = fmaxf(mx, strip[row * JW + ch + c]);
      mx = fmaxf(mx, __shfl_xor(mx, 16));
      float sum = 0.f;
      for (int c = 0; c < 128; ++c) {
        float p = __expf(strip[row * JW + ch + c] - mx);
        strip[row * JW + ch + c] = p;
        sum += p;
      }
      sum += __shfl_xor(sum, 16);
      float inv = 1.0f / sum;
      size_t abase = ((((size_t)bb * NHEAD + h) * NW + w) * WSZ + qi0 + row) * JW;
      for (int c = 0; c < 128; ++c) {
        float p = strip[row * JW + ch + c] * inv;
        strip[row * JW + ch + c] = p;
        attn_out[abase + ch + c] = p;
      }
    }

    // ---- out = attn @ v  (A from LDS, B from transposed v) --------------
    f32x8 o[4] = {};
    for (int j0 = 0; j0 < JW; j0 += 32) {
      bf16x16 af;
      {
        int m = lane & 15;
        int kb8 = (lane & 16) ? 16 : 0;
#pragma unroll
        for (int i = 0; i < 16; ++i) {
          af[i]     = (bf16_t)strip[m * JW + j0 + kb8 + i];
        }
      }
#pragma unroll
      for (int nt = 0; nt < 4; ++nt) {
        int d = nt * 16 + (lane & 15);
        const bf16_t* vp = vt + (bh * HD + d) * NTOK;
        bf16x16 bf_;
#pragma unroll
        for (int i = 0; i < 16; ++i) {
          int t = tw - WSZ + j0 + kh + i;
          bf_[i] = (t >= 0) ? vp[t] : (bf16_t)(-1.0f);
        }
        o[nt] = wmma_bf16(af, bf_, o[nt]);
      }
    }
#pragma unroll
    for (int nt = 0; nt < 4; ++nt) {
      int d = nt * 16 + (lane & 15);
#pragma unroll
      for (int r = 0; r < 8; ++r) {
        int i = qi0 + mh + r;
        ob[((size_t)bb * NTOK + tw + i) * INNER + h * HD + d] = (bf16_t)o[nt][r];
      }
    }
  }
}

// ---------------------------------------------------------------------------
// out = ob(32768x512) * w_fc^T + b_fc ; w_fc stored (DMODEL, INNER) = [n][k].
__global__ __launch_bounds__(128) void la_out_proj(
    const bf16_t* __restrict__ ob, const bf16_t* __restrict__ wfcb,
    const float* __restrict__ bfc, float* __restrict__ out) {
  const int wave = threadIdx.x >> 5;
  const int m0 = blockIdx.x * 64 + wave * 16;
  const int d0 = blockIdx.y * 64;

  f32x8 acc[4] = {};
#pragma unroll
  for (int k0 = 0; k0 < INNER; k0 += 32) {
    bf16x16 a = load_a_frag(ob + (size_t)m0 * INNER + k0, INNER);
#pragma unroll
    for (int nt = 0; nt < 4; ++nt) {
      bf16x16 b = load_b_frag_nk(wfcb + (size_t)(d0 + nt * 16) * INNER + k0, INNER);
      acc[nt] = wmma_bf16(a, b, acc[nt]);
    }
  }
  const int lane = threadIdx.x & 31;
  const int mh = (lane & 16) ? 8 : 0;
#pragma unroll
  for (int nt = 0; nt < 4; ++nt) {
    int d = d0 + nt * 16 + (lane & 15);
    float bias = bfc[d];
#pragma unroll
    for (int r = 0; r < 8; ++r) {
      int tokg = m0 + mh + r;
      out[(size_t)tokg * DMODEL + d] = acc[nt][r] + bias;
    }
  }
}

// ---------------------------------------------------------------------------
extern "C" void kernel_launch(void* const* d_in, const int* in_sizes, int n_in,
                              void* d_out, int out_size, void* d_ws, size_t ws_size,
                              hipStream_t stream) {
  const float* x  = (const float*)d_in[0];
  const unsigned char* mask = (const unsigned char*)d_in[1];  // bool array
  const float* wq = (const float*)d_in[2];
  const float* wk = (const float*)d_in[3];
  const float* wv = (const float*)d_in[4];
  const float* wfc = (const float*)d_in[5];
  const float* bfc = (const float*)d_in[6];

  float* out  = (float*)d_out;                       // (4,8192,256)
  float* attn = out + (size_t)BATCH * NTOK * DMODEL; // (4,8,64,128,256)

  // workspace carve-up (bf16 intermediates), 256B aligned
  char* ws = (char*)d_ws;
  auto take = [&](size_t bytes) {
    char* p = ws;
    ws += (bytes + 255) & ~(size_t)255;
    return p;
  };
  bf16_t* xb  = (bf16_t*)take((size_t)MROWS * DMODEL * 2);       // 16.8 MB
  bf16_t* qd  = (bf16_t*)take((size_t)BATCH * NHEAD * NTOK * HD * 2);
  bf16_t* kd  = (bf16_t*)take((size_t)BATCH * NHEAD * NTOK * HD * 2);
  bf16_t* vtd = (bf16_t*)take((size_t)BATCH * NHEAD * HD * NTOK * 2);
  bf16_t* obuf= (bf16_t*)take((size_t)MROWS * INNER * 2);
  bf16_t* wqb = (bf16_t*)take((size_t)INNER * DMODEL * 2);
  bf16_t* wkb = (bf16_t*)take((size_t)INNER * DMODEL * 2);
  bf16_t* wvb = (bf16_t*)take((size_t)INNER * DMODEL * 2);
  bf16_t* wfb = (bf16_t*)take((size_t)DMODEL * INNER * 2);

  // 1) pack f32 -> bf16
  {
    int n = MROWS * DMODEL;
    la_cvt_bf16<<<(n + 255) / 256, 256, 0, stream>>>(x, xb, n);
    int nw_ = INNER * DMODEL;
    la_cvt_bf16<<<(nw_ + 255) / 256, 256, 0, stream>>>(wq, wqb, nw_);
    la_cvt_bf16<<<(nw_ + 255) / 256, 256, 0, stream>>>(wk, wkb, nw_);
    la_cvt_bf16<<<(nw_ + 255) / 256, 256, 0, stream>>>(wv, wvb, nw_);
    la_cvt_bf16<<<(nw_ + 255) / 256, 256, 0, stream>>>(wfc, wfb, nw_);
  }

  // 2) QKV projections (WMMA)
  la_qkv_gemm<<<dim3(MROWS / 64, INNER / 64, 3), 128, 0, stream>>>(
      xb, wqb, wkb, wvb, qd, kd, vtd);

  // 3) windowed attention (TDM k-tile staging + WMMA + LDS softmax)
  la_attn<<<BATCH * NHEAD * NW, 128, 0, stream>>>(qd, kd, vtd, mask, attn, obuf);

  // 4) output projection + bias (WMMA)
  la_out_proj<<<dim3(MROWS / 64, DMODEL / 64), 128, 0, stream>>>(
      obuf, wfb, bfc, out);
}
